// PretrainedMoE_44444321579344
// MI455X (gfx1250) — matrix operand order repaired and checked by reading, not compiled, not run on hardware
//
#include <hip/hip_runtime.h>
#include <hip/hip_bf16.h>
#include <math.h>

// Problem constants (match reference)
#define B_   8192
#define D_   768
#define E_   16
#define H_   768
#define C_   100
#define CP_  128      // C padded so N-tile grid is even for register blocking
#define BT_  64       // rows per block tile
#define NWAVE 8       // 256 threads = 8 wave32

typedef __attribute__((ext_vector_type(16))) __bf16 v16bf;
typedef __attribute__((ext_vector_type(8)))  __bf16 v8bf;
typedef __attribute__((ext_vector_type(8)))  float  v8f;

union Frag16 { v16bf v; v8bf h[2]; };
union Acc8   { v8f   v; float f[8]; };

__device__ __forceinline__ float gelu_exact(float x) {
    return 0.5f * x * (1.0f + erff(x * 0.70710678118654752f));
}

// Permuted activation layout in LDS: act[rowLane 16][mtile 4][K]
// logical row rr (0..63) -> slot ((rr & 15) * 4 + (rr >> 4)).
// A lane's 4 A-tile fragments are then at one base pointer + {0,1,2,3}*K*2B —
// all within the 16-bit DS immediate-offset range, so the k-loop needs a single
// pointer increment and zero address VALU (avoids WMMA->VALU hazard NOPs).
__device__ __forceinline__ int perm_slot(int row) {
    return ((row & 15) << 2) + (row >> 4);
}

// A fragment (16x32 bf16): ISA layout — lanes 0-15 hold row M=lane with K 0..7 & 16..23,
// lanes 16-31 hold K 8..15 & 24..31. Halves at +0 and +16 elems from per-lane pointer.
__device__ __forceinline__ v16bf frag_a(const __bf16* p) {
    Frag16 f;
    f.h[0] = *(const v8bf*)(p);
    f.h[1] = *(const v8bf*)(p + 16);
    return f.v;
}

// B fragment (32x16 bf16), weights pre-transposed [n][k]: lane holds column n = lane&15,
// K (lane>>4)*8..+7 & 16+(lane>>4)*8..+7: halves at +0 and +16 elems (matches A).
__device__ __forceinline__ v16bf frag_b(const __bf16* p) {
    Frag16 f;
    f.h[0] = *(const v8bf*)(p);
    f.h[1] = *(const v8bf*)(p + 16);
    return f.v;
}

#define WMMA_BF16(a, b, c) __builtin_amdgcn_wmma_f32_16x16x32_bf16( \
    false, (a), false, (b), (short)0, (c), false, false)

// 4x2 register blocking: a wave owns all 4 M-tiles x 2 N-tiles (64x32 macro-tile).
// Per k-step: 4 A-frag (1 LDS pointer, immediate offsets) + 2 B-frag loads feed 8 WMMAs.
template <typename Epi>
__device__ __forceinline__ void wave_gemm4x2(const __bf16* Abuf, int Kd,
                                             const __bf16* Bg, int ldk,
                                             int Ntiles, int wave, int lane, Epi epi) {
    const int Nmac = Ntiles >> 1;
    const int rl = lane & 15;
    const int kh = lane >> 4;
    const __bf16* aBase = Abuf + (size_t)rl * (Kd << 2) + (kh << 3);
    for (int t = wave; t < Nmac; t += NWAVE) {
        const int nt = t << 1;
        const __bf16* pb = Bg + (size_t)(nt * 16 + rl) * ldk + (kh << 3);
        const __bf16* pa = aBase;
        v8f acc[4][2];
        const v8f vz = {};
#pragma unroll
        for (int i = 0; i < 4; ++i) { acc[i][0] = vz; acc[i][1] = vz; }
#pragma unroll 2
        for (int k0 = 0; k0 < Kd; k0 += 32) {
            v16bf a0 = frag_a(pa);
            v16bf a1 = frag_a(pa + Kd);
            v16bf a2 = frag_a(pa + 2 * Kd);
            v16bf a3 = frag_a(pa + 3 * Kd);
            v16bf b0 = frag_b(pb);
            v16bf b1 = frag_b(pb + 16 * ldk);
            pa += 32;
            pb += 32;
            acc[0][0] = WMMA_BF16(a0, b0, acc[0][0]);
            acc[0][1] = WMMA_BF16(a0, b1, acc[0][1]);
            acc[1][0] = WMMA_BF16(a1, b0, acc[1][0]);
            acc[1][1] = WMMA_BF16(a1, b1, acc[1][1]);
            acc[2][0] = WMMA_BF16(a2, b0, acc[2][0]);
            acc[2][1] = WMMA_BF16(a2, b1, acc[2][1]);
            acc[3][0] = WMMA_BF16(a3, b0, acc[3][0]);
            acc[3][1] = WMMA_BF16(a3, b1, acc[3][1]);
        }
        const int rlo = kh << 3;
        Acc8 u;
#pragma unroll
        for (int i = 0; i < 4; ++i) {
#pragma unroll
            for (int j = 0; j < 2; ++j) {
                u.v = acc[i][j];
                const int rbase = i * 16 + rlo;
                const int col   = (nt + j) * 16 + rl;
#pragma unroll
                for (int r = 0; r < 8; ++r) epi(rbase + r, col, u.f[r]);
            }
        }
    }
}

// 2x2 register blocking for the small GEMM3 (so all 8 waves get a macro-tile).
// A read from the permuted layout with a single pointer + immediate offsets.
template <typename Epi>
__device__ __forceinline__ void wave_gemm2x2(const __bf16* Abuf, int Kd,
                                             const __bf16* Bg, int ldk,
                                             int Mtiles, int Ntiles,
                                             int wave, int lane, Epi epi) {
    const int Nmac  = Ntiles >> 1;
    const int total = (Mtiles >> 1) * Nmac;
    const int rl = lane & 15;
    const int kh = lane >> 4;
    const __bf16* aBase = Abuf + (size_t)rl * (Kd << 2) + (kh << 3);
    for (int t = wave; t < total; t += NWAVE) {
        const int mt = (t / Nmac) << 1;
        const int nt = (t % Nmac) << 1;
        const __bf16* pa = aBase + (size_t)mt * Kd;
        const __bf16* pb = Bg + (size_t)(nt * 16 + rl) * ldk + (kh << 3);
        v8f acc00 = {}, acc01 = {}, acc10 = {}, acc11 = {};
#pragma unroll 4
        for (int k0 = 0; k0 < Kd; k0 += 32) {
            v16bf a0 = frag_a(pa);
            v16bf a1 = frag_a(pa + Kd);
            v16bf b0 = frag_b(pb);
            v16bf b1 = frag_b(pb + 16 * ldk);
            pa += 32;
            pb += 32;
            acc00 = WMMA_BF16(a0, b0, acc00);
            acc01 = WMMA_BF16(a0, b1, acc01);
            acc10 = WMMA_BF16(a1, b0, acc10);
            acc11 = WMMA_BF16(a1, b1, acc11);
        }
        const int rlo = kh << 3;
        Acc8 u;
#pragma unroll
        for (int i = 0; i < 2; ++i) {
#pragma unroll
            for (int j = 0; j < 2; ++j) {
                u.v = (i == 0) ? (j == 0 ? acc00 : acc01) : (j == 0 ? acc10 : acc11);
                const int rbase = (mt + i) * 16 + rlo;
                const int col   = (nt + j) * 16 + rl;
#pragma unroll
                for (int r = 0; r < 8; ++r) epi(rbase + r, col, u.f[r]);
            }
        }
    }
}

// ---- Weight prep: f32 [E][K][N] -> bf16 transposed [E][NP][K] (zero-pad n>=N) ----
__global__ void prep_wT_kernel(const float* __restrict__ src, __bf16* __restrict__ dst,
                               int K, int N, int NP, long total) {
    long i = (long)blockIdx.x * blockDim.x + threadIdx.x;
    if (i >= total) return;
    int k = (int)(i % K);
    long t = i / K;
    int n = (int)(t % NP);
    int e = (int)(t / NP);
    float v = (n < N) ? src[((size_t)e * K + k) * N + n] : 0.0f;
    dst[((size_t)e * NP + n) * K + k] = (__bf16)v;
}

// ---- Router logits (tiny GEMM, plain VALU) ----
__global__ void gate_logits_kernel(const float* __restrict__ feat,
                                   const float* __restrict__ rw,
                                   const float* __restrict__ rb,
                                   float* __restrict__ logits) {
    int t = blockIdx.x * blockDim.x + threadIdx.x;
    if (t >= B_ * E_) return;
    int b = t / E_, e = t % E_;
    float acc = rb[e];
    const float* fr = feat + (size_t)b * D_;
#pragma unroll 4
    for (int d = 0; d < D_; ++d) acc += fr[d] * rw[d * E_ + e];
    logits[t] = acc;
}

// ---- Gate softmax + top-4 (desc order, renormalized) ----
__global__ void gate_topk_kernel(const float* __restrict__ logits,
                                 float* __restrict__ gate_out,
                                 float* __restrict__ tv, int* __restrict__ ti) {
    int b = blockIdx.x * blockDim.x + threadIdx.x;
    if (b >= B_) return;
    float p[E_];
    float mx = -3.0e38f;
#pragma unroll
    for (int e = 0; e < E_; ++e) { p[e] = logits[b * E_ + e]; mx = fmaxf(mx, p[e]); }
    float s = 0.f;
#pragma unroll
    for (int e = 0; e < E_; ++e) { p[e] = __expf(p[e] - mx); s += p[e]; }
    float inv = 1.0f / s;
#pragma unroll
    for (int e = 0; e < E_; ++e) { p[e] *= inv; gate_out[b * E_ + e] = p[e]; }
    float tsum = 0.f; float vals[4]; int idx[4];
#pragma unroll
    for (int k = 0; k < 4; ++k) {
        float best = -1.f; int bi = 0;
#pragma unroll
        for (int e = 0; e < E_; ++e) if (p[e] > best) { best = p[e]; bi = e; }
        vals[k] = best; idx[k] = bi; p[bi] = -2.f; tsum += best;
    }
    float r = 1.0f / tsum;
#pragma unroll
    for (int k = 0; k < 4; ++k) { tv[b * 4 + k] = vals[k] * r; ti[b * 4 + k] = idx[k]; }
}

// ---- Fused per-expert pipeline: LN+affine -> GEMM1+GELU -> GEMM2+GELU -> GEMM3 -> softmax ----
__global__ __launch_bounds__(256) void fused_expert_kernel(
    const float* __restrict__ feat,
    const float* __restrict__ ln_g, const float* __restrict__ ln_b,
    const __bf16* __restrict__ wT1, const float* __restrict__ b1,
    const __bf16* __restrict__ wT2, const float* __restrict__ b2,
    const __bf16* __restrict__ wT3, const float* __restrict__ b3,
    float* __restrict__ all_probs) {
    __shared__ __align__(16) __bf16 actBuf[BT_ * D_];   // 96 KB, permuted [16][4][K]
    __shared__ __align__(16) __bf16 hBuf[BT_ * H_];     // 96 KB, permuted (reused as f32 logits)

    const int bt   = blockIdx.x;
    const int e    = blockIdx.y;
    const int tid  = threadIdx.x;
    const int lane = tid & 31;
    const int wave = tid >> 5;
    const int row0 = bt * BT_;

    // Stage 1: LayerNorm + expert affine -> bf16 permuted tile (4 threads/row, 192 elems each)
    {
        const int r = tid >> 2;
        const int q = tid & 3;
        const float* fr = feat + (size_t)(row0 + r) * D_ + q * 192;
        float s = 0.f, s2 = 0.f;
#pragma unroll 4
        for (int j = 0; j < 192; ++j) { float v = fr[j]; s += v; s2 += v * v; }
        s  += __shfl_xor(s, 1);  s  += __shfl_xor(s, 2);
        s2 += __shfl_xor(s2, 1); s2 += __shfl_xor(s2, 2);
        const float mean = s * (1.0f / D_);
        const float var  = s2 * (1.0f / D_) - mean * mean;
        const float rstd = rsqrtf(var + 1e-5f);
        __bf16* dstRow = actBuf + (size_t)perm_slot(r) * D_;
#pragma unroll 4
        for (int j = 0; j < 192; ++j) {
            int d = q * 192 + j;
            float v = (fr[j] - mean) * rstd;
            dstRow[d] = (__bf16)(v * ln_g[e * D_ + d] + ln_b[e * D_ + d]);
        }
    }
    __syncthreads();

    // GEMM1: [64xD] x [DxH] -> GELU -> hBuf (bf16, permuted)
    wave_gemm4x2(actBuf, D_, wT1 + (size_t)e * H_ * D_, D_, H_ / 16, wave, lane,
                 [&](int rr, int cc, float v) {
                     hBuf[perm_slot(rr) * H_ + cc] = (__bf16)gelu_exact(v + b1[e * H_ + cc]);
                 });
    __syncthreads();

    // GEMM2: [64xH] x [HxH] -> GELU -> actBuf (bf16, permuted, reuse)
    wave_gemm4x2(hBuf, H_, wT2 + (size_t)e * H_ * H_, H_, H_ / 16, wave, lane,
                 [&](int rr, int cc, float v) {
                     actBuf[perm_slot(rr) * H_ + cc] = (__bf16)gelu_exact(v + b2[e * H_ + cc]);
                 });
    __syncthreads();

    // GEMM3: [64xH] x [HxCP] -> logits (f32, plain layout) into hBuf region
    float* logits = (float*)hBuf;
    wave_gemm2x2(actBuf, H_, wT3 + (size_t)e * CP_ * H_, H_, BT_ / 16, CP_ / 16, wave, lane,
                 [&](int rr, int cc, float v) {
                     logits[rr * CP_ + cc] = (cc < C_) ? (v + b3[e * C_ + cc]) : -3.0e38f;
                 });
    __syncthreads();

    // Row softmax over C=100 (4 threads per row, 25 cols each) -> all_probs
    {
        const int r = tid >> 2;
        const int q = tid & 3;
        float* lr = logits + r * CP_ + q * 25;
        float mx = -3.0e38f;
#pragma unroll
        for (int j = 0; j < 25; ++j) mx = fmaxf(mx, lr[j]);
        mx = fmaxf(mx, __shfl_xor(mx, 1));
        mx = fmaxf(mx, __shfl_xor(mx, 2));
        float s = 0.f;
#pragma unroll
        for (int j = 0; j < 25; ++j) { float ev = __expf(lr[j] - mx); lr[j] = ev; s += ev; }
        s += __shfl_xor(s, 1);
        s += __shfl_xor(s, 2);
        const float inv = 1.0f / s;
        float* out = all_probs + ((size_t)e * B_ + (row0 + r)) * C_ + q * 25;
#pragma unroll
        for (int j = 0; j < 25; ++j) out[j] = lr[j] * inv;
    }
}

// ---- market_probs: softmax(selected logits) == all_probs[topk_idx], so just gather ----
__global__ void market_kernel(const float* __restrict__ all_probs,
                              const float* __restrict__ tv, const int* __restrict__ ti,
                              float* __restrict__ market) {
    int t = blockIdx.x * blockDim.x + threadIdx.x;
    if (t >= B_ * C_) return;
    int b = t / C_, c = t % C_;
    float acc = 0.f;
#pragma unroll
    for (int k = 0; k < 4; ++k) {
        int e = ti[b * 4 + k];
        acc += tv[b * 4 + k] * all_probs[((size_t)e * B_ + b) * C_ + c];
    }
    market[t] = acc;
}

extern "C" void kernel_launch(void* const* d_in, const int* in_sizes, int n_in,
                              void* d_out, int out_size, void* d_ws, size_t ws_size,
                              hipStream_t stream) {
    const float* feat = (const float*)d_in[0];
    const float* rw   = (const float*)d_in[1];
    const float* rb   = (const float*)d_in[2];
    const float* lng  = (const float*)d_in[3];
    const float* lnb  = (const float*)d_in[4];
    const float* w1   = (const float*)d_in[5];
    const float* b1   = (const float*)d_in[6];
    const float* w2   = (const float*)d_in[7];
    const float* b2   = (const float*)d_in[8];
    const float* w3   = (const float*)d_in[9];
    const float* b3   = (const float*)d_in[10];

    float* market     = (float*)d_out;                       // [B, C]
    float* all_probs  = market + (size_t)B_ * C_;            // [E, B, C]
    float* gate_probs = all_probs + (size_t)E_ * B_ * C_;    // [B, E]

    // Workspace layout (~41.8 MB): bf16 transposed weights + router scratch
    char* ws = (char*)d_ws;
    size_t o = 0;
    __bf16* wT1 = (__bf16*)(ws + o); o += (size_t)E_ * H_ * D_ * 2;
    __bf16* wT2 = (__bf16*)(ws + o); o += (size_t)E_ * H_ * H_ * 2;
    __bf16* wT3 = (__bf16*)(ws + o); o += (size_t)E_ * CP_ * H_ * 2;
    float*  glog = (float*)(ws + o); o += (size_t)B_ * E_ * 4;
    float*  tv   = (float*)(ws + o); o += (size_t)B_ * 4 * 4;
    int*    ti   = (int*)(ws + o);   o += (size_t)B_ * 4 * 4;
    (void)ws_size; (void)in_sizes; (void)n_in; (void)out_size;

    {
        long tot = (long)E_ * H_ * D_;
        prep_wT_kernel<<<(tot + 255) / 256, 256, 0, stream>>>(w1, wT1, D_, H_, H_, tot);
    }
    {
        long tot = (long)E_ * H_ * H_;
        prep_wT_kernel<<<(tot + 255) / 256, 256, 0, stream>>>(w2, wT2, H_, H_, H_, tot);
    }
    {
        long tot = (long)E_ * CP_ * H_;
        prep_wT_kernel<<<(tot + 255) / 256, 256, 0, stream>>>(w3, wT3, H_, C_, CP_, tot);
    }

    gate_logits_kernel<<<(B_ * E_ + 255) / 256, 256, 0, stream>>>(feat, rw, rb, glog);
    gate_topk_kernel<<<(B_ + 255) / 256, 256, 0, stream>>>(glog, gate_probs, tv, ti);

    fused_expert_kernel<<<dim3(B_ / BT_, E_), 256, 0, stream>>>(
        feat, lng, lnb, wT1, b1, wT2, b2, wT3, b3, all_probs);

    market_kernel<<<(B_ * C_ + 255) / 256, 256, 0, stream>>>(all_probs, tv, ti, market);
}
